// GCNEModel_89541478187026
// MI455X (gfx1250) — matrix operand order repaired and checked by reading, not compiled, not run on hardware
//
#include <hip/hip_runtime.h>
#include <cmath>

// ---------------- problem constants (from reference) ----------------
#define N_GENES 15135
#define N_FEAT  4
#define PE_DIM  32
#define KIN1    (N_FEAT + PE_DIM)   // 36
#define HID     64
#define BS      8
#define N_EDGES 242160
#define HID_FC  256
#define M_ROWS  (BS * N_GENES)      // 121080; row r = n*8 + b (node-major)

typedef __attribute__((ext_vector_type(16))) _Float16 v16h;
typedef __attribute__((ext_vector_type(8)))  _Float16 v8h;
typedef __attribute__((ext_vector_type(8)))  float    v8f;

// ---------------- small setup kernels ----------------
__global__ void k_init(float* __restrict__ g, float* __restrict__ deg) {
    int i = blockIdx.x * blockDim.x + threadIdx.x;
    if (i < M_ROWS) g[i] = 0.f;          // fc accumulator
    if (i < N_GENES) deg[i] = 1.f;       // self-loop weight
}

__global__ void k_degree(const int* __restrict__ dst, float* __restrict__ deg) {
    int e = blockIdx.x * blockDim.x + threadIdx.x;
    if (e < N_EDGES) atomicAdd(&deg[dst[e]], 1.f);
}

__global__ void k_dinv(float* __restrict__ deg) {
    int n = blockIdx.x * blockDim.x + threadIdx.x;
    if (n < N_GENES) deg[n] = rsqrtf(deg[n]);   // in-place deg -> dinv
}

__global__ void k_zero4(float* __restrict__ p, int count4) {
    int i = blockIdx.x * blockDim.x + threadIdx.x;
    if (i < count4) ((float4*)p)[i] = make_float4(0.f, 0.f, 0.f, 0.f);
}

// ---- stage layer-0 input: xin16[r][k] = concat(x, peW^T+peb) zero-padded to 64, f16
__global__ __launch_bounds__(256) void k_stage(
    const float* __restrict__ x, const float* __restrict__ pelW,
    const float* __restrict__ pelb, _Float16* __restrict__ xin)
{
    int t = blockIdx.x * 256 + threadIdx.x;         // over M_ROWS*64
    if (t >= M_ROWS * HID) return;
    int r = t >> 6, k = t & 63;
    int n = r >> 3, b = r & 7;
    float v = 0.f;
    if (k < N_FEAT)
        v = x[(size_t)(b * N_GENES + n) * N_FEAT + k];
    else if (k < KIN1)
        v = pelW[(size_t)(k - N_FEAT) * N_GENES + n] + pelb[k - N_FEAT];
    xin[t] = (_Float16)v;
}

// ---------------- WMMA GEMM core helpers ----------------
// Per-wave 16x16 tile; 8 waves/block -> 32 rows x 64 cols.
// A frag (ISA 7.12.2, 16-bit A 16x32): lane half 0 -> K {0..7,16..23}, half 1 -> +8.
// B frag: lane half 0 -> K {0..7,16..23}, half 1 -> +8; col = lane%16.
__device__ __forceinline__ v16h load_bfrag(const float* __restrict__ W,
                                           int col, int kc, int bbase)
{
    const float4* pb = (const float4*)(W + col * HID + kc * 32 + bbase / 2);
    float4 b0 = pb[0], b1 = pb[1], b2 = pb[4], b3 = pb[5];
    v16h bf;
    bf[0]=(_Float16)b0.x; bf[1]=(_Float16)b0.y; bf[2]=(_Float16)b0.z; bf[3]=(_Float16)b0.w;
    bf[4]=(_Float16)b1.x; bf[5]=(_Float16)b1.y; bf[6]=(_Float16)b1.z; bf[7]=(_Float16)b1.w;
    bf[8]=(_Float16)b2.x; bf[9]=(_Float16)b2.y; bf[10]=(_Float16)b2.z; bf[11]=(_Float16)b2.w;
    bf[12]=(_Float16)b3.x; bf[13]=(_Float16)b3.y; bf[14]=(_Float16)b3.z; bf[15]=(_Float16)b3.w;
    return bf;
}

// Store epilogue: validity of the 8-row group is uniform (mbase, M_ROWS both %8==0),
// so a single branch guards 8 unconditional immediate-offset stores.
__device__ __forceinline__ void store_ctile(float* __restrict__ out,
                                            int mbase, int col, const v8f& c)
{
    if (mbase < M_ROWS) {
        float* p = out + (size_t)mbase * HID + col;
        #pragma unroll
        for (int v = 0; v < 8; ++v) p[v * HID] = c[v];
    }
}

// GEMM, A already f16 (layer 0, staged & zero-padded to K=64)
__global__ __launch_bounds__(256) void k_gemm16(
    const _Float16* __restrict__ in, const float* __restrict__ W,
    float* __restrict__ out)
{
    const int lane = threadIdx.x & 31;
    const int wv   = threadIdx.x >> 5;
    const int rt   = blockIdx.x * 2 + (wv >> 2);
    const int ct   = wv & 3;
    const int lrow = lane & 15;
    const int hi   = lane >> 4;
    const int row  = rt * 16 + lrow;
    const int col  = ct * 16 + lrow;
    const int abase = hi * 8, bbase = hi * 16;

    const int rowc = row < M_ROWS ? row : (M_ROWS - 1);   // clamp; masked at store
    const _Float16* pa = in + (size_t)rowc * HID;

    v8f c = {};
    #pragma unroll
    for (int kc = 0; kc < 2; ++kc) {
        v8h lo  = *(const v8h*)(pa + kc * 32 + abase);        // K s0..s0+7
        v8h hi8 = *(const v8h*)(pa + kc * 32 + abase + 16);   // K s0+16..s0+23
        v16h af = __builtin_shufflevector(lo, hi8,
                    0,1,2,3,4,5,6,7,8,9,10,11,12,13,14,15);
        v16h bf = load_bfrag(W, col, kc, bbase);
        c = __builtin_amdgcn_wmma_f32_16x16x32_f16(
                false, af, false, bf, (short)0, c, false, false);
    }

    store_ctile(out, rt * 16 + hi * 8, col, c);
}

// GEMM, A in f32 (layers 1..2, K = 64)
__global__ __launch_bounds__(256) void k_gemm32(
    const float* __restrict__ in, const float* __restrict__ W,
    float* __restrict__ out)
{
    const int lane = threadIdx.x & 31;
    const int wv   = threadIdx.x >> 5;
    const int rt   = blockIdx.x * 2 + (wv >> 2);
    const int ct   = wv & 3;
    const int lrow = lane & 15;
    const int hi   = lane >> 4;
    const int row  = rt * 16 + lrow;
    const int col  = ct * 16 + lrow;
    const int abase = hi * 8, bbase = hi * 16;

    const int rowc = row < M_ROWS ? row : (M_ROWS - 1);
    const float* pa = in + (size_t)rowc * HID;

    v8f c = {};
    #pragma unroll
    for (int kc = 0; kc < 2; ++kc) {
        const float4* p0 = (const float4*)(pa + kc * 32 + abase);
        const float4* p1 = (const float4*)(pa + kc * 32 + abase + 16);
        float4 a0 = p0[0], a1 = p0[1], a2 = p1[0], a3 = p1[1];
        v16h af;
        af[0]=(_Float16)a0.x; af[1]=(_Float16)a0.y; af[2]=(_Float16)a0.z; af[3]=(_Float16)a0.w;
        af[4]=(_Float16)a1.x; af[5]=(_Float16)a1.y; af[6]=(_Float16)a1.z; af[7]=(_Float16)a1.w;
        af[8]=(_Float16)a2.x; af[9]=(_Float16)a2.y; af[10]=(_Float16)a2.z; af[11]=(_Float16)a2.w;
        af[12]=(_Float16)a3.x; af[13]=(_Float16)a3.y; af[14]=(_Float16)a3.z; af[15]=(_Float16)a3.w;
        v16h bf = load_bfrag(W, col, kc, bbase);
        c = __builtin_amdgcn_wmma_f32_16x16x32_f16(
                false, af, false, bf, (short)0, c, false, false);
    }

    store_ctile(out, rt * 16 + hi * 8, col, c);
}

// ---------------- edge scatter: agg[:,dst,:] += h[:,src,:] * dinv[s]*dinv[d] ----
// 2 edges per 256-thread block; rows src*8..src*8+7 contiguous (r = n*8+b).
__global__ __launch_bounds__(256) void k_scatter(
    const int* __restrict__ src, const int* __restrict__ dst,
    const float* __restrict__ dinv,
    const float* __restrict__ h, float* __restrict__ agg)
{
    int t  = threadIdx.x;
    int e  = blockIdx.x * 2 + (t >> 7);
    int tt = t & 127;
    int s = src[e], d = dst[e];
    float norm = dinv[s] * dinv[d];
    int b  = tt >> 4;                               // batch 0..7
    int c4 = tt & 15;                               // float4 chunk 0..15
    const float4* hp = (const float4*)(h + (size_t)(s * 8 + b) * HID);
    float4 v = hp[c4];
    float* ap = agg + (size_t)(d * 8 + b) * HID + c4 * 4;
    atomicAdd(ap + 0, v.x * norm);
    atomicAdd(ap + 1, v.y * norm);
    atomicAdd(ap + 2, v.z * norm);
    atomicAdd(ap + 3, v.w * norm);
}

// ---------------- finalize layer: a = relu(agg + h*dinv^2 + b); fold fc into g ----
__global__ __launch_bounds__(256) void k_finalize(
    const float* __restrict__ h, float* __restrict__ agg /* becomes layer output */,
    const float* __restrict__ dinv, const float* __restrict__ bias,
    const float* __restrict__ fcW, int layer, float* __restrict__ g)
{
    __shared__ float red[256];
    int t = threadIdx.x;
    int r = blockIdx.x * 4 + (t >> 6);              // 4 rows per block
    int f = t & 63;
    int n = r >> 3, b = r & 7;
    float di = dinv[n];
    size_t idx = (size_t)r * HID + f;
    float val = agg[idx] + h[idx] * di * di + bias[f];
    val = fmaxf(val, 0.f);
    agg[idx] = val;
    red[t] = val * fcW[f * 3 + layer];              // xcat[b,n,f*3+layer]*fc_W
    __syncthreads();
    for (int s = 32; s > 0; s >>= 1) {
        if ((t & 63) < s) red[t] += red[t + s];
        __syncthreads();
    }
    if ((t & 63) == 0) g[b * N_GENES + n] += red[t];
}

// ---------------- dense head ----------------
__global__ __launch_bounds__(256) void k_head1(
    const float* __restrict__ g, const float* __restrict__ fcb,
    const float* __restrict__ l1W, const float* __restrict__ l1b,
    float* __restrict__ z)
{
    __shared__ float red[256];
    int t = threadIdx.x;
    int b = blockIdx.x >> 8;
    int j = blockIdx.x & 255;
    float fb = fcb[0];
    float acc = 0.f;
    for (int n = t; n < N_GENES; n += 256)
        acc += (g[b * N_GENES + n] + fb) * l1W[(size_t)j * N_GENES + n];
    red[t] = acc;
    __syncthreads();
    for (int s = 128; s > 0; s >>= 1) {
        if (t < s) red[t] += red[t + s];
        __syncthreads();
    }
    if (t == 0) z[b * HID_FC + j] = fmaxf(red[0] + l1b[j], 0.f);
}

__global__ void k_head2(const float* __restrict__ z,
                        const float* __restrict__ l2W, const float* __restrict__ l2b,
                        float* __restrict__ out)
{
    __shared__ float lg[16];
    int t = threadIdx.x;                            // 32 threads
    if (t < 16) {
        int b = t >> 1, c = t & 1;
        float acc = l2b[c];
        for (int k = 0; k < HID_FC; ++k)
            acc += z[b * HID_FC + k] * l2W[c * HID_FC + k];
        lg[t] = acc;
    }
    __syncthreads();
    if (t < 16) {
        int b = t >> 1;
        float a0 = lg[b * 2], a1 = lg[b * 2 + 1];
        float m = fmaxf(a0, a1);
        float lse = m + logf(expf(a0 - m) + expf(a1 - m));
        out[t] = lg[t] - lse;                       // log_softmax, [8,2] row-major
    }
}

// ---------------- driver ----------------
extern "C" void kernel_launch(void* const* d_in, const int* in_sizes, int n_in,
                              void* d_out, int out_size, void* d_ws, size_t ws_size,
                              hipStream_t stream)
{
    const float* x    = (const float*)d_in[0];
    const int*   eidx = (const int*)d_in[2];       // [2, E]
    const float* pelW = (const float*)d_in[3];
    const float* pelb = (const float*)d_in[4];
    const float* c1W  = (const float*)d_in[5];  const float* c1b = (const float*)d_in[6];
    const float* c2W  = (const float*)d_in[7];  const float* c2b = (const float*)d_in[8];
    const float* c3W  = (const float*)d_in[9];  const float* c3b = (const float*)d_in[10];
    const float* fcW  = (const float*)d_in[11]; const float* fcb = (const float*)d_in[12];
    const float* l1W  = (const float*)d_in[13]; const float* l1b = (const float*)d_in[14];
    const float* l2W  = (const float*)d_in[15]; const float* l2b = (const float*)d_in[16];
    float* out = (float*)d_out;

    const int* esrc = eidx;
    const int* edst = eidx + N_EDGES;

    // workspace layout (~78.5 MB total)
    char* ws = (char*)d_ws;
    const size_t HB = (size_t)M_ROWS * HID * sizeof(float);          // 30.99 MB
    float*     bufA = (float*)ws;                                    // h (pre-agg)
    float*     bufB = (float*)(ws + HB);                             // agg -> layer out
    _Float16*  xin  = (_Float16*)(ws + 2 * HB);                      // staged layer-0 A
    char*      tail = ws + 2 * HB + (size_t)M_ROWS * HID * sizeof(_Float16);
    float*     dinv = (float*)tail;                                  // deg -> dinv
    float*     g    = (float*)(tail + ((N_GENES * 4 + 255) & ~255));
    float*     z    = g + M_ROWS;

    k_init  <<<(M_ROWS + 255) / 256, 256, 0, stream>>>(g, dinv);
    k_degree<<<(N_EDGES + 255) / 256, 256, 0, stream>>>(edst, dinv);
    k_dinv  <<<(N_GENES + 255) / 256, 256, 0, stream>>>(dinv);
    k_stage <<<(M_ROWS * HID + 255) / 256, 256, 0, stream>>>(x, pelW, pelb, xin);

    const int rowTiles = (M_ROWS + 15) / 16;                         // 7568
    const int gemmGrid = (rowTiles + 1) / 2;                         // 3784
    const int zero4    = (M_ROWS * HID) / 4;

    const float* convW[3] = {c1W, c2W, c3W};
    const float* convB[3] = {c1b, c2b, c3b};
    for (int l = 0; l < 3; ++l) {
        if (l == 0)
            k_gemm16<<<gemmGrid, 256, 0, stream>>>(xin, convW[0], bufA);
        else
            k_gemm32<<<gemmGrid, 256, 0, stream>>>(bufB, convW[l], bufA);
        k_zero4   <<<(zero4 + 255) / 256, 256, 0, stream>>>(bufB, zero4);
        k_scatter <<<N_EDGES / 2, 256, 0, stream>>>(esrc, edst, dinv, bufA, bufB);
        k_finalize<<<M_ROWS / 4, 256, 0, stream>>>(bufA, bufB, dinv, convB[l],
                                                   fcW, l, g);
    }

    k_head1<<<BS * HID_FC, 256, 0, stream>>>(g, fcb, l1W, l1b, z);
    k_head2<<<1, 32, 0, stream>>>(z, l2W, l2b, out);
}
